// ECO_GROW_86526411145583
// MI455X (gfx1250) — matrix operand order repaired and checked by reading, not compiled, not run on hardware
//
#include <hip/hip_runtime.h>
#include <hip/hip_bf16.h>
#include <math.h>

typedef _Float16 half_t;
typedef __attribute__((ext_vector_type(16))) _Float16 v16h;
typedef __attribute__((ext_vector_type(8)))  _Float16 v8h;
typedef __attribute__((ext_vector_type(8)))  float    v8f;

namespace {
constexpr int NN   = 297;    // nodes
constexpr int FDIM = 3;      // node feat dim
constexpr int CH   = 256;    // channels
constexpr int HID  = 2048;   // hidden
constexpr int RH   = 256;    // rnn hidden
constexpr int NG   = 5;      // graph_cnt
constexpr int NT   = 20;     // NG + lstm_k
constexpr int NL   = 3;      // lstm layers
constexpr int NE   = 9600;   // edges per graph
constexpr int NEL  = 20000;  // label edges
constexpr int NP   = 320;    // nodes padded to multiple of 32
constexpr int FK   = NG*CH + RH + FDIM; // 1539
constexpr int FKP  = 1568;   // FK padded to multiple of 32
constexpr float BN_EPS = 1e-5f;
enum { EP_BIAS = 1, EP_RELU = 2, EP_ACCUM = 4 };
}

// ---------------------------------------------------------------------------
// f16 WMMA GEMM, fp32 accumulate.  C[M,Nn] (+)= A[M,K] * B[K,Nn].
//   A  : f16 row-major [M][K]      (lda = K, multiple of 32)
//   BT : f16 transposed [Nn][K]    (ldb = K, multiple of 32)
// M % 16 == 0, Nn % 64 == 0, K % 32 == 0 guaranteed by caller => no guards,
// EXEC is full everywhere, and each fragment is contiguous 128-bit loads.
// 4 waves/block, each wave owns one 16x16 tile (block tile 16x64).
// ---------------------------------------------------------------------------
__global__ __launch_bounds__(128) void gemm_f16_wmma(
    const half_t* __restrict__ A, int lda,
    const half_t* __restrict__ BT, int ldb,
    const float* __restrict__ bias,
    float* __restrict__ C, int ldc,
    int K, int flags)
{
  const int lane = threadIdx.x & 31;
  const int wave = threadIdx.x >> 5;
  const int hi   = lane >> 4;    // half-wave select
  const int lo   = lane & 15;    // row (A) / col (B,C) within tile
  const int m0   = blockIdx.y * 16;
  const int n0   = blockIdx.x * 64 + wave * 16;

  // A frag: lane row = lo, covers k = hi*16+[0..7] and
  // k = hi*16+8+[0..7]  -> two contiguous 16B chunks.
  const half_t* pa = A + (size_t)(m0 + lo) * (size_t)lda + hi * 16;
  // B frag: lane col = lo, k = hi*16 + [0..15] -> one contiguous 32B chunk.
  const half_t* pb = BT + (size_t)(n0 + lo) * (size_t)ldb + hi * 16;

  v8f acc = {0.f,0.f,0.f,0.f,0.f,0.f,0.f,0.f};

  for (int k0 = 0; k0 < K; k0 += 32) {
    __builtin_prefetch(pa + 32, 0, 1);
    __builtin_prefetch(pb + 32, 0, 1);
    v8h  a0 = *(const v8h*)(pa);
    v8h  a1 = *(const v8h*)(pa + 8);
    v16h b  = *(const v16h*)(pb);
    v16h a  = __builtin_shufflevector(a0, a1, 0,1,2,3,4,5,6,7,8,9,10,11,12,13,14,15);
    acc = __builtin_amdgcn_wmma_f32_16x16x32_f16(
        /*neg_a=*/false, a, /*neg_b=*/false, b,
        /*c_mod=*/(short)0, acc, /*reuse_a=*/false, /*reuse_b=*/false);
    pa += 32;
    pb += 32;
  }

  // C/D layout: lane = col lo, row = v + 8*hi.
  const int n = n0 + lo;
  float* cp = C + (size_t)(m0 + 8 * hi) * (size_t)ldc + n;
  const float bv = (flags & EP_BIAS) ? bias[n] : 0.f;
#pragma unroll
  for (int v = 0; v < 8; ++v) {
    float r = acc[v] + bv;
    if (flags & EP_ACCUM) r += cp[(size_t)v * ldc];
    if (flags & EP_RELU)  r = fmaxf(r, 0.f);
    cp[(size_t)v * ldc] = r;
  }
}

// ---------------------------------------------------------------------------
// Conversion / layout kernels
// ---------------------------------------------------------------------------
__global__ void zero_f32_kernel(float* p, int n) {
  int i = blockIdx.x * blockDim.x + threadIdx.x;
  if (i < n) p[i] = 0.f;
}
__global__ void zero_f16_kernel(half_t* p, int n) {
  int i = blockIdx.x * blockDim.x + threadIdx.x;
  if (i < n) p[i] = (half_t)0.f;
}
// straight f32 -> f16
__global__ void cvt_kernel(const float* __restrict__ s, half_t* __restrict__ d, int n) {
  int i = blockIdx.x * blockDim.x + threadIdx.x;
  if (i < n) d[i] = (half_t)s[i];
}
// transpose + convert: src f32 [K][Nn] row-major -> dst f16 [Nn][Kp], zero pad k>=K
__global__ void cvtT_kernel(const float* __restrict__ s, half_t* __restrict__ d,
                            int K, int Nn, int Kp) {
  int i = blockIdx.x * blockDim.x + threadIdx.x;
  if (i >= Nn * Kp) return;
  int k = i % Kp, n = i / Kp;
  d[i] = (half_t)((k < K) ? s[(size_t)k * Nn + n] : 0.f);
}

// ---------------------------------------------------------------------------
// Model-specific kernels
// ---------------------------------------------------------------------------
// x_emb[t][n][c] = x_all[t][n][:] @ W0 + b0 (zero on padded rows)
__global__ void embed_kernel(const float* __restrict__ gx, const float* __restrict__ tx,
                             const float* __restrict__ W0, const float* __restrict__ b0,
                             float* __restrict__ xe) {
  int idx = blockIdx.x * blockDim.x + threadIdx.x;
  if (idx >= NT * NP * CH) return;
  int c = idx % CH, n = (idx / CH) % NP, t = idx / (CH * NP);
  float r = 0.f;
  if (n < NN) {
    const float* x = (t < NG) ? &gx[((size_t)t * NN + n) * FDIM]
                              : &tx[((size_t)(t - NG) * NN + n) * FDIM];
    r = b0[c];
#pragma unroll
    for (int f = 0; f < FDIM; ++f) r += x[f] * W0[f * CH + c];
  }
  xe[idx] = r;
}

// to_dense_adj scatter-add: adj[t][src][dst] += w (adj pre-zeroed)
__global__ void adj_scatter_kernel(const int* __restrict__ gei, const int* __restrict__ tei,
                                   const float* __restrict__ gew, const float* __restrict__ tew,
                                   float* __restrict__ adj) {
  int idx = blockIdx.x * blockDim.x + threadIdx.x;
  if (idx >= NT * NE) return;
  int e = idx % NE, t = idx / NE;
  const int* ei = (t < NG) ? &gei[(size_t)t * 2 * NE] : &tei[(size_t)(t - NG) * 2 * NE];
  float w = (t < NG) ? gew[(size_t)t * NE + e] : tew[(size_t)(t - NG) * NE + e];
  int s = ei[e], d = ei[NE + e];
  atomicAdd(&adj[((size_t)t * NP + s) * NP + d], w);
}

// Per (t,c): mean/rstd over the 297 real nodes. LDS tree reduction.
__global__ void bn_stats_kernel(const float* __restrict__ zb,
                                float* __restrict__ mu, float* __restrict__ rstd) {
  int t = blockIdx.y, c = blockIdx.x;
  __shared__ float s1[64], s2[64];
  float a = 0.f, b = 0.f;
  for (int n = threadIdx.x; n < NN; n += 64) {
    float v = zb[((size_t)t * NP + n) * CH + c];
    a += v; b += v * v;
  }
  s1[threadIdx.x] = a; s2[threadIdx.x] = b;
  __syncthreads();
  for (int s = 32; s > 0; s >>= 1) {
    if (threadIdx.x < (unsigned)s) {
      s1[threadIdx.x] += s1[threadIdx.x + s];
      s2[threadIdx.x] += s2[threadIdx.x + s];
    }
    __syncthreads();
  }
  if (threadIdx.x == 0) {
    float m = s1[0] / (float)NN;
    float var = s2[0] / (float)NN - m * m;
    mu[t * CH + c]   = m;
    rstd[t * CH + c] = rsqrtf(var + BN_EPS);
  }
}

// zb -> normalized (in place); seq = z_all + residual (spatial graphs only)
__global__ void bn_apply_kernel(float* __restrict__ zb, const float* __restrict__ xe,
                                const float* __restrict__ mu, const float* __restrict__ rstd,
                                const float* __restrict__ gamma, const float* __restrict__ beta,
                                float* __restrict__ seq) {
  int idx = blockIdx.x * blockDim.x + threadIdx.x;
  if (idx >= NT * NP * CH) return;
  int c = idx % CH, n = (idx / CH) % NP, t = idx / (CH * NP);
  float z = 0.f;
  if (n < NN)
    z = (zb[idx] - mu[t * CH + c]) * rstd[t * CH + c] * gamma[t * CH + c] + beta[t * CH + c];
  zb[idx] = z;
  float s = z;
  if (t < NG) s += xe[idx];   // xe padded rows are zero
  seq[idx] = s;
}

__global__ void bias_sum_kernel(const float* a, const float* b, float* o, int n) {
  int i = blockIdx.x * blockDim.x + threadIdx.x;
  if (i < n) o[i] = a[i] + b[i];
}

__device__ __forceinline__ float sigm(float x) { return 1.f / (1.f + expf(-x)); }

// gates[n][0:4RH] (i,f,g,o chunks) -> update c (f32), h (f16, feeds next GEMM),
// and emit h into the f16 sequence buffer for the next layer.
__global__ void lstm_point_kernel(const float* __restrict__ gates,
                                  half_t* __restrict__ h16, float* __restrict__ c,
                                  half_t* __restrict__ outseq16) {
  int idx = blockIdx.x * blockDim.x + threadIdx.x;
  if (idx >= NN * RH) return;
  int j = idx % RH, n = idx / RH;
  const float* g = &gates[(size_t)n * 4 * RH];
  float i_ = sigm(g[j]);
  float f_ = sigm(g[RH + j]);
  float gg = tanhf(g[2 * RH + j]);
  float o_ = sigm(g[3 * RH + j]);
  float cc = f_ * c[(size_t)n * RH + j] + i_ * gg;
  float hh = o_ * tanhf(cc);
  c[(size_t)n * RH + j] = cc;
  h16[(size_t)n * RH + j] = (half_t)hh;
  outseq16[(size_t)n * RH + j] = (half_t)hh;
}

// feat16[n] = [softmax(scores)_g * z_all[g][n][:]] ++ rnn_last[n] ++ node_feature[n],
// written directly in f16, zero on pad rows/cols (this is lin1's A operand).
__global__ void feat_kernel(const float* __restrict__ z_all, const half_t* __restrict__ rnn_last,
                            const float* __restrict__ nf, const float* __restrict__ scores,
                            half_t* __restrict__ feat16) {
  int idx = blockIdx.x * blockDim.x + threadIdx.x;
  if (idx >= NP * FKP) return;
  int c = idx % FKP, n = idx / FKP;
  float r = 0.f;
  if (n < NN && c < FK) {
    if (c < NG * CH) {
      int g = c / CH, cc = c % CH;
      float mx = scores[0];
#pragma unroll
      for (int i = 1; i < NG; ++i) mx = fmaxf(mx, scores[i]);
      float den = 0.f;
#pragma unroll
      for (int i = 0; i < NG; ++i) den += expf(scores[i] - mx);
      float w = expf(scores[g] - mx) / den;
      r = w * z_all[((size_t)g * NP + n) * CH + cc];
    } else if (c < NG * CH + RH) {
      r = (float)rnn_last[(size_t)n * RH + (c - NG * CH)];
    } else {
      r = nf[(size_t)n * FDIM + (c - NG * CH - RH)];
    }
  }
  feat16[idx] = (half_t)r;
}

// copy padded [NP][CH] rows < NN into d_out `out` region (contiguous)
__global__ void copy_out_kernel(const float* __restrict__ s, float* __restrict__ d) {
  int i = blockIdx.x * blockDim.x + threadIdx.x;
  if (i < NN * CH) d[i] = s[i];
}

// p1 = sigmoid(out@reg1_W + b), p3 = sigmoid(out@reg3_W + b)
__global__ void heads_kernel(const float* __restrict__ out,
                             const float* __restrict__ r1W, const float* __restrict__ r1b,
                             const float* __restrict__ r3W, const float* __restrict__ r3b,
                             float* __restrict__ p1, float* __restrict__ p3) {
  int n = blockIdx.x * blockDim.x + threadIdx.x;
  if (n >= NN) return;
  float a = r1b[0], b = r3b[0];
  for (int c = 0; c < CH; ++c) {
    float v = out[(size_t)n * CH + c];
    a += v * r1W[c];
    b += v * r3W[c];
  }
  p1[n] = sigm(a);
  p3[n] = sigm(b);
}

// r[e] = dot(out[src], out[dst])
__global__ void edge_kernel(const float* __restrict__ out, const int* __restrict__ eli,
                            float* __restrict__ r) {
  int e = blockIdx.x * blockDim.x + threadIdx.x;
  if (e >= NEL) return;
  int s = eli[e], d = eli[NEL + e];
  float acc = 0.f;
  for (int c = 0; c < CH; ++c)
    acc += out[(size_t)s * CH + c] * out[(size_t)d * CH + c];
  r[e] = acc;
}

// ---------------------------------------------------------------------------
// Host-side orchestration
// ---------------------------------------------------------------------------
static inline void launch_gemm(hipStream_t s, const half_t* A, int lda,
                               const half_t* BT, int ldb, const float* bias,
                               float* C, int ldc, int M, int Nn, int K, int flags) {
  dim3 grid(Nn / 64, M / 16);
  gemm_f16_wmma<<<grid, 128, 0, s>>>(A, lda, BT, ldb, bias, C, ldc, K, flags);
}
static inline void launch_zero_f32(hipStream_t s, float* p, int n) {
  zero_f32_kernel<<<(n + 255) / 256, 256, 0, s>>>(p, n);
}
static inline void launch_zero_f16(hipStream_t s, half_t* p, int n) {
  zero_f16_kernel<<<(n + 255) / 256, 256, 0, s>>>(p, n);
}
static inline void launch_cvt(hipStream_t s, const float* src, half_t* dst, int n) {
  cvt_kernel<<<(n + 255) / 256, 256, 0, s>>>(src, dst, n);
}
static inline void launch_cvtT(hipStream_t s, const float* src, half_t* dst,
                               int K, int Nn, int Kp) {
  cvtT_kernel<<<(Nn * Kp + 255) / 256, 256, 0, s>>>(src, dst, K, Nn, Kp);
}

extern "C" void kernel_launch(void* const* d_in, const int* in_sizes, int n_in,
                              void* d_out, int out_size, void* d_ws, size_t ws_size,
                              hipStream_t stream) {
  (void)in_sizes; (void)n_in; (void)out_size; (void)ws_size;

  const float* node_feature = (const float*)d_in[0];
  const float* graphs_x     = (const float*)d_in[1];
  const float* graphs_ew    = (const float*)d_in[2];
  const float* time_x       = (const float*)d_in[3];
  const float* time_ew      = (const float*)d_in[4];
  const float* W0           = (const float*)d_in[5];
  const float* b0           = (const float*)d_in[6];
  const float* conv1_W      = (const float*)d_in[7];
  const float* conv1_b      = (const float*)d_in[8];
  const float* conv2_W      = (const float*)d_in[9];
  const float* conv2_b      = (const float*)d_in[10];
  const float* bn_gamma     = (const float*)d_in[11];
  const float* bn_beta      = (const float*)d_in[12];
  const float* lstm_W_ih    = (const float*)d_in[13];
  const float* lstm_W_hh    = (const float*)d_in[14];
  const float* lstm_b_ih    = (const float*)d_in[15];
  const float* lstm_b_hh    = (const float*)d_in[16];
  const float* lin1_W       = (const float*)d_in[17];
  const float* lin1_b       = (const float*)d_in[18];
  const float* lin2_W       = (const float*)d_in[19];
  const float* lin2_b       = (const float*)d_in[20];
  const float* reg1_W       = (const float*)d_in[21];
  const float* reg1_b       = (const float*)d_in[22];
  const float* reg3_W       = (const float*)d_in[23];
  const float* reg3_b       = (const float*)d_in[24];
  const float* graph_scores = (const float*)d_in[25];
  const int*   gei          = (const int*)d_in[26];
  const int*   tei          = (const int*)d_in[27];
  const int*   eli          = (const int*)d_in[28];

  // Output layout: p1 [297] | r [20000] | p3 [297] | out [297*256]
  float* out_p1 = (float*)d_out;
  float* out_r  = out_p1 + NN;
  float* out_p3 = out_r + NEL;
  float* out_o  = out_p3 + NN;

  // Workspace carve-up (64B aligned slices)
  char*  wb  = (char*)d_ws;
  size_t off = 0;
  auto carve = [&](size_t bytes) -> void* {
    void* p = wb + off;
    off += (bytes + 63) & ~(size_t)63;
    return p;
  };
  // f32
  float* xe    = (float*)carve(sizeof(float) * NT * NP * CH);
  float* adj   = (float*)carve(sizeof(float) * NT * NP * NP);
  float* zb    = (float*)carve(sizeof(float) * NT * NP * CH);
  float* seq   = (float*)carve(sizeof(float) * NT * NP * CH);
  float* t1    = (float*)carve(sizeof(float) * NP * HID);
  float* t2    = (float*)carve(sizeof(float) * NP * HID);
  float* mu    = (float*)carve(sizeof(float) * NT * CH);
  float* rstd  = (float*)carve(sizeof(float) * NT * CH);
  float* gates = (float*)carve(sizeof(float) * NP * 4 * RH);
  float* cst   = (float*)carve(sizeof(float) * NP * RH);
  float* bbuf  = (float*)carve(sizeof(float) * NL * 4 * RH);
  float* o1    = (float*)carve(sizeof(float) * NP * 2 * CH);
  float* outb  = (float*)carve(sizeof(float) * NP * CH);
  // f16
  half_t* xe16   = (half_t*)carve(sizeof(half_t) * NT * NP * CH);
  half_t* adj16  = (half_t*)carve(sizeof(half_t) * NT * NP * NP);
  half_t* seq16  = (half_t*)carve(sizeof(half_t) * NT * NP * CH);
  half_t* w1T16  = (half_t*)carve(sizeof(half_t) * HID * CH);   // per-t reuse
  half_t* w2T16  = (half_t*)carve(sizeof(half_t) * CH * HID);   // per-t reuse
  half_t* t1T16  = (half_t*)carve(sizeof(half_t) * HID * NP);
  half_t* t2_16  = (half_t*)carve(sizeof(half_t) * NP * HID);
  half_t* tb16   = (half_t*)carve(sizeof(half_t) * CH * NP);
  half_t* wih16  = (half_t*)carve(sizeof(half_t) * NL * 4 * RH * CH);
  half_t* whh16  = (half_t*)carve(sizeof(half_t) * NL * 4 * RH * RH);
  half_t* h16    = (half_t*)carve(sizeof(half_t) * NP * RH);
  half_t* bufP16 = (half_t*)carve(sizeof(half_t) * NT * NP * RH);
  half_t* bufQ16 = (half_t*)carve(sizeof(half_t) * NT * NP * RH);
  half_t* feat16 = (half_t*)carve(sizeof(half_t) * NP * FKP);
  half_t* l1T16  = (half_t*)carve(sizeof(half_t) * 2 * CH * FKP);
  half_t* l2T16  = (half_t*)carve(sizeof(half_t) * CH * 2 * CH);
  half_t* o1_16  = (half_t*)carve(sizeof(half_t) * NP * 2 * CH);

  // 0) one-time prep: LSTM weights are already [N][K]-layout -> straight cvt;
  //    head weights are [K][N] -> transpose+convert (lin1 zero-padded to 1568).
  launch_cvt(stream, lstm_W_ih, wih16, NL * 4 * RH * CH);
  launch_cvt(stream, lstm_W_hh, whh16, NL * 4 * RH * RH);
  launch_cvtT(stream, lin1_W, l1T16, FK, 2 * CH, FKP);
  launch_cvtT(stream, lin2_W, l2T16, 2 * CH, CH, 2 * CH);
  bias_sum_kernel<<<(NL * 4 * RH + 255) / 256, 256, 0, stream>>>(
      lstm_b_ih, lstm_b_hh, bbuf, NL * 4 * RH);
  launch_zero_f16(stream, bufP16, NT * NP * RH);
  launch_zero_f16(stream, bufQ16, NT * NP * RH);

  // 1) embedding + dense adjacency, converted to f16 operands
  launch_zero_f32(stream, adj, NT * NP * NP);
  embed_kernel<<<(NT * NP * CH + 255) / 256, 256, 0, stream>>>(graphs_x, time_x, W0, b0, xe);
  adj_scatter_kernel<<<(NT * NE + 255) / 256, 256, 0, stream>>>(gei, tei, graphs_ew, time_ew, adj);
  launch_cvt(stream, xe, xe16, NT * NP * CH);
  launch_cvt(stream, adj, adj16, NT * NP * NP);

  // 2) per-graph GCN encoder: 4 WMMA GEMMs each
  for (int t = 0; t < NT; ++t) {
    const half_t* adjt = adj16 + (size_t)t * NP * NP;
    // t1 = x_emb @ W1                     [NP,HID]
    launch_cvtT(stream, conv1_W + (size_t)t * CH * HID, w1T16, CH, HID, CH);
    launch_gemm(stream, xe16 + (size_t)t * NP * CH, CH, w1T16, CH, nullptr,
                t1, HID, NP, HID, CH, 0);
    // t2 = relu(adj @ t1 + b1)            [NP,HID]
    launch_cvtT(stream, t1, t1T16, NP, HID, NP);
    launch_gemm(stream, adjt, NP, t1T16, NP, conv1_b + (size_t)t * HID,
                t2, HID, NP, HID, NP, EP_BIAS | EP_RELU);
    // t1 = t2 @ W2                        [NP,CH]
    launch_cvt(stream, t2, t2_16, NP * HID);
    launch_cvtT(stream, conv2_W + (size_t)t * HID * CH, w2T16, HID, CH, HID);
    launch_gemm(stream, t2_16, HID, w2T16, HID, nullptr, t1, CH, NP, CH, HID, 0);
    // zb[t] = adj @ t1 + b2               [NP,CH]
    launch_cvtT(stream, t1, tb16, NP, CH, NP);
    launch_gemm(stream, adjt, NP, tb16, NP, conv2_b + (size_t)t * CH,
                zb + (size_t)t * NP * CH, CH, NP, CH, NP, EP_BIAS);
  }

  // 3) BatchNorm (batch stats over nodes) + residual into LSTM input
  {
    dim3 g(CH, NT);
    bn_stats_kernel<<<g, 64, 0, stream>>>(zb, mu, rstd);
  }
  bn_apply_kernel<<<(NT * NP * CH + 255) / 256, 256, 0, stream>>>(
      zb, xe, mu, rstd, bn_gamma, bn_beta, seq);
  launch_cvt(stream, seq, seq16, NT * NP * CH);

  // 4) 3-layer LSTM over T=20 steps, batch = nodes; 2 WMMA GEMMs per step
  const half_t* lin = seq16;
  half_t* louts[NL] = {bufP16, bufQ16, bufP16};
  for (int l = 0; l < NL; ++l) {
    launch_zero_f16(stream, h16, NP * RH);
    launch_zero_f32(stream, cst, NP * RH);
    const half_t* Wih = wih16 + (size_t)l * 4 * RH * CH;
    const half_t* Whh = whh16 + (size_t)l * 4 * RH * RH;
    for (int t = 0; t < NT; ++t) {
      // gates = x_t @ Wih^T + (b_ih + b_hh)
      launch_gemm(stream, lin + (size_t)t * NP * CH, CH, Wih, CH,
                  bbuf + (size_t)l * 4 * RH, gates, 4 * RH, NP, 4 * RH, CH, EP_BIAS);
      // gates += h @ Whh^T
      launch_gemm(stream, h16, RH, Whh, RH, nullptr,
                  gates, 4 * RH, NP, 4 * RH, RH, EP_ACCUM);
      lstm_point_kernel<<<(NN * RH + 255) / 256, 256, 0, stream>>>(
          gates, h16, cst, louts[l] + (size_t)t * NP * RH);
    }
    lin = louts[l];
  }
  const half_t* rnn_last = bufP16 + (size_t)(NT - 1) * NP * RH;

  // 5) feature concat + MLP head (WMMA GEMMs)
  feat_kernel<<<(NP * FKP + 255) / 256, 256, 0, stream>>>(
      zb, rnn_last, node_feature, graph_scores, feat16);
  launch_gemm(stream, feat16, FKP, l1T16, FKP, lin1_b,
              o1, 2 * CH, NP, 2 * CH, FKP, EP_BIAS | EP_RELU);
  launch_cvt(stream, o1, o1_16, NP * 2 * CH);
  launch_gemm(stream, o1_16, 2 * CH, l2T16, 2 * CH, lin2_b,
              outb, CH, NP, CH, 2 * CH, EP_BIAS);
  copy_out_kernel<<<(NN * CH + 255) / 256, 256, 0, stream>>>(outb, out_o);

  // 6) decode
  heads_kernel<<<(NN + 255) / 256, 256, 0, stream>>>(
      out_o, reg1_W, reg1_b, reg3_W, reg3_b, out_p1, out_p3);
  edge_kernel<<<(NEL + 255) / 256, 256, 0, stream>>>(out_o, eli, out_r);
}
